// GroupBillinear_51247549776179
// MI455X (gfx1250) — matrix operand order repaired and checked by reading, not compiled, not run on hardware
//
#include <hip/hip_runtime.h>

#define NGROUP   32
#define CHN      512
#define PDIM     16
#define PP       256
#define BATCH    32
#define WDIM     56
#define SPATIAL  (WDIM*WDIM)        /* 3136 */
#define NTOT     (BATCH*SPATIAL)    /* 100352 */
#define ROWS     16
#define NBLK     (NTOT/ROWS)        /* 6272 */
#define TS       64
#define BN_EPS   1e-5f
#define CPADH    (CHN+8)            /* ushort row pad: 1040B stride, 16B aligned */

typedef __attribute__((ext_vector_type(16))) __bf16        v16bf;
typedef __attribute__((ext_vector_type(8)))  float         v8f;
typedef __attribute__((ext_vector_type(4)))  unsigned int  u32x4;
typedef __attribute__((ext_vector_type(8)))  int           i32x8;
typedef __attribute__((ext_vector_type(4)))  int           i32x4;

union frag_u { uint4 q[2]; v16bf v; };

__device__ __forceinline__ unsigned short bf2us(__bf16 b) {
    union { __bf16 b; unsigned short s; } o; o.b = b; return o.s;
}

// ---------------------------------------------------------------------------
// TDM: load a [tile_dim1=512 rows] x [tile_dim0=16 floats] tile (row stride
// SPATIAL elements) from global memory into LDS at lds_off. D# per ISA 8.3/8.4.
// ---------------------------------------------------------------------------
__device__ __forceinline__ void tdm_load_tile_f32(const float* gptr, unsigned lds_off)
{
    const unsigned long long ga = (unsigned long long)(uintptr_t)gptr;
    u32x4 g0;
    g0[0] = 1u;                                              // count=1 (valid, user)
    g0[1] = lds_off;                                         // lds_addr (bytes)
    g0[2] = (unsigned)(ga & 0xFFFFFFFFu);                    // global_addr[31:0]
    g0[3] = (unsigned)((ga >> 32) & 0x1FFFFFFu) | (2u << 30);// global_addr[56:32] | type=2
    i32x8 g1;
    g1[0] = 0x00020000;            // workgroup_mask=0 | data_size=2 (4 bytes)
    g1[1] = (int)(16u << 16);      // tensor_dim0 = 16        (bits 79:48 lo)
    g1[2] = (int)(512u << 16);     // tensor_dim0 hi=0 | tensor_dim1 = 512
    g1[3] = (int)(16u << 16);      // tensor_dim1 hi=0 | tile_dim0 = 16
    g1[4] = 512;                   // tile_dim1 = 512 | tile_dim2 = 0
    g1[5] = SPATIAL;               // tensor_dim0_stride = 3136 (elements)
    g1[6] = 0;                     // stride0 hi | tensor_dim1_stride lo
    g1[7] = 0;
    i32x4 z4 = {0, 0, 0, 0};
#if __has_include(<hip/amd_detail/amd_gfx1250_TDM.h>)
    i32x8 z8 = {0, 0, 0, 0, 0, 0, 0, 0};
    __builtin_amdgcn_tensor_load_to_lds(g0, g1, z4, z4, z8, 0);
#else
    __builtin_amdgcn_tensor_load_to_lds(g0, g1, z4, z4, 0);
#endif
}

// ---------------------------------------------------------------------------
// Kernel W: one-time fc_w f32 -> bf16 (hardware cvt), 4 elements/thread.
// ---------------------------------------------------------------------------
__global__ __launch_bounds__(256)
void conv_w(const float* __restrict__ w, unsigned short* __restrict__ wb)
{
    const int i = (blockIdx.x * 256 + threadIdx.x) * 4;
    const float4 f = *(const float4*)(w + i);
    ushort4 o;
    o.x = bf2us((__bf16)f.x);
    o.y = bf2us((__bf16)f.y);
    o.z = bf2us((__bf16)f.z);
    o.w = bf2us((__bf16)f.w);
    *(ushort4*)(wb + i) = o;
}

// ---------------------------------------------------------------------------
// Kernel A: TDM-staged x tile -> bf16 GEMM (t = x + x@W^T + b) -> per-row
// 16x16x32 Gram -> tanh -> s-buffer + deterministic BN partials.
// ---------------------------------------------------------------------------
__global__ __launch_bounds__(256)
void fused_gemm_gram(const float* __restrict__ x,
                     const unsigned short* __restrict__ fc_wb,
                     const float* __restrict__ fc_b,
                     float* __restrict__ s_buf, float* __restrict__ psum,
                     float* __restrict__ psq)
{
    __shared__ float          x_stage[CHN][ROWS];   // 32 KB, TDM dest: [channel][r]
    __shared__ unsigned short a_bf [ROWS][CPADH];   // 16.25 KB, A in row-major bf16
    __shared__ unsigned short t_tr [ROWS][CPADH];   // 16.25 KB, t as [r][p*32+g]
    __shared__ float          s_tile[ROWS][PP];     // 16 KB

    const int wg   = blockIdx.x;
    const int n0   = wg * ROWS;
    const int bimg = n0 / SPATIAL;       // 3136 % 16 == 0 -> tile within one image
    const int sp0  = n0 % SPATIAL;
    const int tid  = threadIdx.x;

    // Phase 1: async-tensor DMA of the NHWC gather (512 rows of 16 floats).
    if (tid < 32) {
        tdm_load_tile_f32(x + (size_t)bimg * CHN * SPATIAL + sp0,
                          (unsigned)(uintptr_t)&x_stage[0][0]);
        __builtin_amdgcn_s_wait_tensorcnt(0);
    }
    __syncthreads();

    // Phase 1b: single bf16 conversion of the A tile (transposed read).
    {
        const int r  = tid & 15;
        const int c0 = tid >> 4;
        #pragma unroll 4
        for (int i = 0; i < CHN / 16; ++i) {
            const int c = c0 + i * 16;
            a_bf[r][c] = bf2us((__bf16)x_stage[c][r]);
        }
    }
    __syncthreads();

    const int wave = tid >> 5;
    const int lane = tid & 31;
    const int mcol = lane & 15;
    const int kbA  = (lane < 16) ? 0 : 8;    // A fragment K base (pair layout)
    const int kbB  = (lane < 16) ? 0 : 16;   // B fragment K base (half-wave split)

    // Phase 2: GEMM. Each wave owns 4 column tiles (64 output channels).
    v8f acc[4] = {};
    for (int kc = 0; kc < CHN / 32; ++kc) {
        frag_u fa;
        fa.q[0] = *(const uint4*)&a_bf[mcol][kc * 32 + kbA];
        fa.q[1] = *(const uint4*)&a_bf[mcol][kc * 32 + kbA + 16];
        #pragma unroll
        for (int ci = 0; ci < 4; ++ci) {
            const int ct      = wave * 4 + ci;
            const int row_out = ct * 16 + mcol;          // B[k,n] = fc_w[n,k]
            const unsigned short* bp = fc_wb + (size_t)row_out * CHN + kc * 32 + kbA;
            if (kc + 1 < CHN / 32)
                __builtin_prefetch(bp + 32, 0, 1);       // global_prefetch_b8
            frag_u fb;
            fb.q[0] = ((const uint4*)bp)[0];
            fb.q[1] = ((const uint4*)bp)[2];
            acc[ci] = __builtin_amdgcn_wmma_f32_16x16x32_bf16(
                false, fa.v, false, fb.v, (short)0, acc[ci], false, false);
        }
    }

    // Epilogue: t = residual + gemm + bias; store into Gram-fragment layout.
    #pragma unroll
    for (int ci = 0; ci < 4; ++ci) {
        const int ct  = wave * 4 + ci;
        const int col = ct * 16 + mcol;
        const float bi = fc_b[col];
        #pragma unroll
        for (int j = 0; j < 8; ++j) {
            const int r = j + ((lane < 16) ? 0 : 8);   // C/D: M=j (+8 for hi half)
            const float tv = acc[ci][j] + x_stage[col][r] + bi;
            t_tr[r][mcol * 32 + ct] = bf2us((__bf16)tv);   // [r][p*32 + g]
        }
    }
    __syncthreads();

    // Phase 3: per-row Gram S = A^T A over 32 groups == one 16x16x32 WMMA.
    #pragma unroll
    for (int rr0 = 0; rr0 < 2; ++rr0) {
        const int r = wave * 2 + rr0;
        const unsigned short* base = &t_tr[r][mcol * 32];
        frag_u fa, fb;
        fa.q[0] = *(const uint4*)(base + kbA);          // A[p,g] = t[g*16+p]
        fa.q[1] = *(const uint4*)(base + kbA + 16);
        fb.q[0] = *(const uint4*)(base + kbA);          // B[g,q] = t[g*16+q]
        fb.q[1] = *(const uint4*)(base + kbA + 16);
        v8f s = {};
        s = __builtin_amdgcn_wmma_f32_16x16x32_bf16(
                false, fa.v, false, fb.v, (short)0, s, false, false);
        #pragma unroll
        for (int j = 0; j < 8; ++j) {
            const int p = j + ((lane < 16) ? 0 : 8);
            s_tile[r][p * 16 + mcol] = tanhf(s[j] * (1.0f / 32.0f));
        }
    }
    __syncthreads();

    // Phase 4: coalesced s-buffer write + fixed-order (deterministic) partials.
    {
        const int j = tid;                // 0..255 == (p,q) flat index
        float sum = 0.f, sq = 0.f;
        #pragma unroll
        for (int r = 0; r < ROWS; ++r) {
            const float v = s_tile[r][j];
            sum += v; sq += v * v;
            s_buf[(size_t)(n0 + r) * PP + j] = v;
        }
        psum[(size_t)wg * PP + j] = sum;
        psq [(size_t)wg * PP + j] = sq;
    }
}

// ---------------------------------------------------------------------------
// Kernel B: deterministic tree reduction -> mean/rstd per s-channel.
// ---------------------------------------------------------------------------
__global__ __launch_bounds__(256)
void bn_stats(const float* __restrict__ psum, const float* __restrict__ psq,
              float* __restrict__ stats)
{
    __shared__ float red0[256];
    __shared__ float red1[256];
    const int j = blockIdx.x;             // 0..255
    const int t = threadIdx.x;
    float s = 0.f, q = 0.f;
    for (int wg = t; wg < NBLK; wg += 256) {
        s += psum[(size_t)wg * PP + j];
        q += psq [(size_t)wg * PP + j];
    }
    red0[t] = s; red1[t] = q;
    __syncthreads();
    for (int off = 128; off > 0; off >>= 1) {
        if (t < off) { red0[t] += red0[t + off]; red1[t] += red1[t + off]; }
        __syncthreads();
    }
    if (t == 0) {
        const float inv_n = 1.0f / (float)NTOT;
        const float mean  = red0[0] * inv_n;
        const float var   = red1[0] * inv_n - mean * mean;
        stats[j]      = mean;
        stats[PP + j] = rsqrtf(var + BN_EPS);
    }
}

// ---------------------------------------------------------------------------
// Kernel C: out = x + BN(upsample(s)), s staged in padded LDS.
// ---------------------------------------------------------------------------
__global__ __launch_bounds__(256)
void bn_apply(const float* __restrict__ x, const float* __restrict__ s_buf,
              const float* __restrict__ stats, const float* __restrict__ gamma,
              const float* __restrict__ beta, float* __restrict__ out)
{
    __shared__ float s_lds[TS * 257];     // 257 pad -> bank-conflict free
    const int blk  = blockIdx.x;
    const int bimg = blk / (SPATIAL / TS);
    const int spb  = (blk % (SPATIAL / TS)) * TS;
    const int nb   = bimg * SPATIAL + spb;
    const int tid  = threadIdx.x;

    for (int i = tid; i < TS * PP; i += 256) {
        const int r = i >> 8, jj = i & 255;
        s_lds[r * 257 + jj] = s_buf[(size_t)(nb + r) * PP + jj];
    }
    __syncthreads();

    const int sp_l = tid & (TS - 1);
    const int coff = tid >> 6;            // 0..3
    const size_t base = (size_t)bimg * CHN * SPATIAL + spb + sp_l;
    for (int c = coff; c < CHN; c += 4) {
        const int jj     = c >> 1;        // nearest-neighbor 256 -> 512
        const float mean = stats[jj];
        const float rstd = stats[PP + jj];
        const float g    = gamma[c];
        const float bt   = beta[c];
        const size_t idx = base + (size_t)c * SPATIAL;
        const float sv   = s_lds[sp_l * 257 + jj];
        out[idx] = x[idx] + (sv - mean) * rstd * g + bt;
    }
}

// ---------------------------------------------------------------------------
extern "C" void kernel_launch(void* const* d_in, const int* in_sizes, int n_in,
                              void* d_out, int out_size, void* d_ws, size_t ws_size,
                              hipStream_t stream)
{
    const float* x     = (const float*)d_in[0];
    const float* fc_w  = (const float*)d_in[1];
    const float* fc_b  = (const float*)d_in[2];
    const float* gamma = (const float*)d_in[3];
    const float* beta  = (const float*)d_in[4];
    float* out = (float*)d_out;

    float* s_buf = (float*)d_ws;                       // NTOT*PP floats (~103 MB)
    float* psum  = s_buf + (size_t)NTOT * PP;          // NBLK*PP
    float* psq   = psum  + (size_t)NBLK * PP;          // NBLK*PP
    float* stats = psq   + (size_t)NBLK * PP;          // 512
    unsigned short* fc_wb = (unsigned short*)(stats + 512);  // CHN*CHN bf16 (512 KB)

    conv_w<<<(CHN * CHN) / (256 * 4), 256, 0, stream>>>(fc_w, fc_wb);
    fused_gemm_gram<<<NBLK, 256, 0, stream>>>(x, fc_wb, fc_b, s_buf, psum, psq);
    bn_stats<<<PP, 256, 0, stream>>>(psum, psq, stats);
    bn_apply<<<BATCH * (SPATIAL / TS), 256, 0, stream>>>(x, s_buf, stats, gamma, beta, out);
}
